// EncodingLayer_28243704939344
// MI455X (gfx1250) — matrix operand order repaired and checked by reading, not compiled, not run on hardware
//
#include <hip/hip_runtime.h>
#include <hip/hip_bf16.h>

typedef __attribute__((ext_vector_type(16))) __bf16 v16bf;
typedef __attribute__((ext_vector_type(8)))  float  v8f;

__device__ __forceinline__ __bf16 f2bf(float f) {
    return (__bf16)f;   // native f32->bf16 convert (v_cvt_*_bf16_f32)
}

__device__ __forceinline__ float sigmoidf(float x) {
    return 1.0f / (1.0f + expf(-x));
}

// ---------------- degree / normalization ----------------
__global__ void k_set1(float* __restrict__ deg, int N) {
    int i = blockIdx.x * blockDim.x + threadIdx.x;
    if (i < N) deg[i] = 1.0f;                 // self-loop weight
}

__global__ void k_deg_edges(float* __restrict__ deg, const int* __restrict__ dst,
                            const float* __restrict__ w, int E) {
    int e = blockIdx.x * blockDim.x + threadIdx.x;
    if (e < E) atomicAdd(&deg[dst[e]], w[e]);
}

__global__ void k_rsqrt(float* __restrict__ d, int N) {
    int i = blockIdx.x * blockDim.x + threadIdx.x;
    if (i < N) {
        float v = d[i];
        d[i] = (v > 0.0f) ? rsqrtf(v) : 0.0f;
    }
}

// ---------------- WMMA GEMM:  Y[N x F] = act(X[N x 128] @ W[128 x F] + bias) ----
// blockDim = 128 (4 waves); each wave owns a 16-row stripe, all F columns.
// W is pre-swizzled into LDS as ready-made 32-byte B fragments:
//   fragment(kkIdx, ct, lane) = v16bf where elem j = bf16(W[kk + (j<8 ? sel*8 + j : 16 + sel*8 + j-8)][ct*16 + n15])
//   (sel = lane>>4, n15 = lane&15) -> one contiguous 32B LDS read per WMMA operand.
template <int F, int ACT>  // ACT: 0 = none, 1 = tanh
__global__ __launch_bounds__(128)
void k_gemm(const float* __restrict__ X, const float* __restrict__ W,
            const float* __restrict__ bias, float* __restrict__ Y, int N) {
    constexpr int CT = F / 16;               // column tiles
    __shared__ v16bf ldsB[4 * CT * 32];      // 4 K-chunks * CT tiles * 32 lanes

    const int tid = threadIdx.x;
    // stage + swizzle weights (K=128 x F row-major) into B-fragment layout
    {
        __bf16* wb = (__bf16*)ldsB;
        const float4* w4 = (const float4*)W;
        const int total4 = (128 * F) / 4;
        for (int idx = tid; idx < total4; idx += 128) {
            float4 v = w4[idx];
            int o   = idx * 4;
            int k   = o / F;
            int col0 = o - k * F;
            int kkIdx = k >> 5;
            int k5    = k & 31;
            int sel   = (k5 >> 3) & 1;
            int j     = (k5 & 7) | ((k5 >> 4) << 3);
            float vv[4] = {v.x, v.y, v.z, v.w};
#pragma unroll
            for (int c = 0; c < 4; ++c) {
                int col  = col0 + c;
                int ct   = col >> 4;
                int n15  = col & 15;
                int lane = sel * 16 + n15;
                wb[(((kkIdx * CT + ct) * 32) + lane) * 16 + j] = f2bf(vv[c]);
            }
        }
    }
    __syncthreads();

    const int lane = tid & 31;
    const int wave = tid >> 5;
    const int n15  = lane & 15;
    const int sel  = lane >> 4;
    const int rowBase = blockIdx.x * 64 + wave * 16;

    int loadRow = rowBase + n15;
    if (loadRow >= N) loadRow = N - 1;        // clamp: keep EXEC uniform for WMMA
    const float* xrow = X + (size_t)loadRow * 128;

    v8f acc[CT];
    const v8f zero = {0.f, 0.f, 0.f, 0.f, 0.f, 0.f, 0.f, 0.f};
#pragma unroll
    for (int t = 0; t < CT; ++t) acc[t] = zero;

#pragma unroll
    for (int kkIdx = 0; kkIdx < 4; ++kkIdx) {
        const int kk = kkIdx * 32;
        // A fragment: 16-bit A 16x32 layout
        // lanes 0-15: elems 0..7 -> K kk+0..7,  elems 8..15 -> K kk+16..23
        // lanes 16-31: elems 0..7 -> K kk+8..15, elems 8..15 -> K kk+24..31
        v16bf a;
        {
            const float4* p0 = (const float4*)(xrow + kk + sel * 8);
            const float4* p1 = (const float4*)(xrow + kk + 16 + sel * 8);
            float4 x0 = p0[0], x1 = p0[1], x2 = p1[0], x3 = p1[1];
            a[0]  = f2bf(x0.x); a[1]  = f2bf(x0.y); a[2]  = f2bf(x0.z); a[3]  = f2bf(x0.w);
            a[4]  = f2bf(x1.x); a[5]  = f2bf(x1.y); a[6]  = f2bf(x1.z); a[7]  = f2bf(x1.w);
            a[8]  = f2bf(x2.x); a[9]  = f2bf(x2.y); a[10] = f2bf(x2.z); a[11] = f2bf(x2.w);
            a[12] = f2bf(x3.x); a[13] = f2bf(x3.y); a[14] = f2bf(x3.z); a[15] = f2bf(x3.w);
        }
#pragma unroll
        for (int ct = 0; ct < CT; ++ct) {
            v16bf b = ldsB[(kkIdx * CT + ct) * 32 + lane];   // 32B contiguous -> 2x ds_load_b128
            acc[ct] = __builtin_amdgcn_wmma_f32_16x16x32_bf16(
                false, a, false, b, (short)0, acc[ct], false, false);
        }
    }

    // C layout: elem r -> row rowBase + sel*8 + r, col = ct*16 + n15
#pragma unroll
    for (int ct = 0; ct < CT; ++ct) {
        const int col = ct * 16 + n15;
        const float bv = bias ? bias[col] : 0.0f;
#pragma unroll
        for (int r = 0; r < 8; ++r) {
            const int row = rowBase + sel * 8 + r;
            if (row < N) {
                float v = acc[ct][r] + bv;
                if (ACT == 1) v = tanhf(v);
                Y[(size_t)row * F + col] = v;
            }
        }
    }
}

// ---------------- aggregation ----------------
// self-loop term initializes agg: agg[i,:] = h[i,:] * dinv[i]^2
__global__ void k_self(const float* __restrict__ h, const float* __restrict__ dinv,
                       float* __restrict__ agg, int N, int F) {
    long long t = (long long)blockIdx.x * blockDim.x + threadIdx.x;
    const int perRow = F >> 2;
    long long total = (long long)N * perRow;
    if (t >= total) return;
    int i  = (int)(t / perRow);
    int f0 = (int)(t % perRow) * 4;
    float d = dinv[i];
    float s = d * d;
    float4 hv = *(const float4*)(h + (size_t)i * F + f0);
    float4 o;
    o.x = hv.x * s; o.y = hv.y * s; o.z = hv.z * s; o.w = hv.w * s;
    *(float4*)(agg + (size_t)i * F + f0) = o;
}

// per-edge scatter: agg[dst,:] += h[src,:] * dinv[src]*w*dinv[dst]
__global__ void k_edges(const float* __restrict__ h, const float* __restrict__ dinv,
                        const int* __restrict__ src, const int* __restrict__ dst,
                        const float* __restrict__ w, float* __restrict__ agg,
                        int E, int F) {
    long long t = (long long)blockIdx.x * blockDim.x + threadIdx.x;
    const int perEdge = F >> 2;
    long long total = (long long)E * perEdge;
    if (t >= total) return;
    int e  = (int)(t / perEdge);
    int f0 = (int)(t % perEdge) * 4;
    int s  = src[e];
    int d  = dst[e];
    float nrm = dinv[s] * w[e] * dinv[d];
    float4 hv = *(const float4*)(h + (size_t)s * F + f0);
    float* out = agg + (size_t)d * F + f0;
    atomicAdd(out + 0, hv.x * nrm);
    atomicAdd(out + 1, hv.y * nrm);
    atomicAdd(out + 2, hv.z * nrm);
    atomicAdd(out + 3, hv.w * nrm);
}

__global__ void k_bias_tanh(const float* __restrict__ agg, const float* __restrict__ b,
                            float* __restrict__ out, long long total, int F) {
    long long t = (long long)blockIdx.x * blockDim.x + threadIdx.x;
    if (t >= total) return;
    int f = (int)(t % F);
    out[t] = tanhf(agg[t] + b[f]);
}

// enc[:,0:64] = sigmoid(agg2 + b2)
__global__ void k_enc_adj(const float* __restrict__ agg2, const float* __restrict__ b2,
                          float* __restrict__ enc, int N) {
    long long t = (long long)blockIdx.x * blockDim.x + threadIdx.x;
    long long total = (long long)N * 64;
    if (t >= total) return;
    int i = (int)(t >> 6);
    int f = (int)(t & 63);
    enc[(size_t)i * 128 + f] = sigmoidf(agg2[t] + b2[f]);
}

// enc[:,64:96] = sigmoid(gdv @ gdv_W + gdv_b)
__global__ void k_enc_gdv(const float* __restrict__ gdv, const float* __restrict__ Wg,
                          const float* __restrict__ bg, float* __restrict__ enc, int N) {
    long long t = (long long)blockIdx.x * blockDim.x + threadIdx.x;
    long long total = (long long)N * 32;
    if (t >= total) return;
    int i = (int)(t >> 5);
    int j = (int)(t & 31);
    float acc = bg[j];
    const float* g = gdv + (size_t)i * 73;
#pragma unroll 1
    for (int k = 0; k < 73; ++k) acc += g[k] * Wg[k * 32 + j];
    enc[(size_t)i * 128 + 64 + j] = sigmoidf(acc);
}

// enc[:,96:128] = sigmoid(pr @ pr_W + pr_b)
__global__ void k_enc_pr(const float* __restrict__ pr, const float* __restrict__ Wp,
                         const float* __restrict__ bp, float* __restrict__ enc, int N) {
    long long t = (long long)blockIdx.x * blockDim.x + threadIdx.x;
    long long total = (long long)N * 32;
    if (t >= total) return;
    int i = (int)(t >> 5);
    int j = (int)(t & 31);
    enc[(size_t)i * 128 + 96 + j] = sigmoidf(pr[i] * Wp[j] + bp[j]);
}

static inline unsigned gdiv(long long a, int b) { return (unsigned)((a + b - 1) / b); }

extern "C" void kernel_launch(void* const* d_in, const int* in_sizes, int n_in,
                              void* d_out, int out_size, void* d_ws, size_t ws_size,
                              hipStream_t stream) {
    const float* feat  = (const float*)d_in[0];
    const float* gdv   = (const float*)d_in[1];
    const float* pr    = (const float*)d_in[2];
    const int*   eidx  = (const int*)d_in[3];
    const float* ew    = (const float*)d_in[4];
    const float* W1    = (const float*)d_in[5];
    const float* b1    = (const float*)d_in[6];
    const float* W2    = (const float*)d_in[7];
    const float* b2    = (const float*)d_in[8];
    const float* gdvW  = (const float*)d_in[9];
    const float* gdvB  = (const float*)d_in[10];
    const float* prW   = (const float*)d_in[11];
    const float* prB   = (const float*)d_in[12];
    const float* encW1 = (const float*)d_in[13];
    const float* encB1 = (const float*)d_in[14];
    const float* encW2 = (const float*)d_in[15];
    const float* encB2 = (const float*)d_in[16];

    const int N = in_sizes[0] / 128;
    const int E = in_sizes[4];
    const int* src = eidx;
    const int* dst = eidx + E;

    // workspace: dinv[Npad] | A[N*128] | B[N*128]
    size_t nPad = ((size_t)N + 255) & ~(size_t)255;
    float* dinv = (float*)d_ws;
    float* A = dinv + nPad;
    float* B = A + (size_t)N * 128;

    float* out = (float*)d_out;
    const unsigned gB = (unsigned)((N + 63) / 64);  // GEMM blocks (64 rows each)

    // 1) degrees -> dinv
    k_set1<<<gdiv(N, 256), 256, 0, stream>>>(dinv, N);
    k_deg_edges<<<gdiv(E, 256), 256, 0, stream>>>(dinv, dst, ew, E);
    k_rsqrt<<<gdiv(N, 256), 256, 0, stream>>>(dinv, N);

    // 2) layer 1: h = feat @ W1 ; aggregate ; tanh(+b1)
    k_gemm<128, 0><<<gB, 128, 0, stream>>>(feat, W1, nullptr, A, N);
    k_self<<<gdiv((long long)N * 32, 256), 256, 0, stream>>>(A, dinv, B, N, 128);
    k_edges<<<gdiv((long long)E * 32, 256), 256, 0, stream>>>(A, dinv, src, dst, ew, B, E, 128);
    k_bias_tanh<<<gdiv((long long)N * 128, 256), 256, 0, stream>>>(B, b1, A, (long long)N * 128, 128);

    // 3) layer 2: h2 = h @ W2 ; aggregate ; sigmoid(+b2) -> enc[:,0:64]
    k_gemm<64, 0><<<gB, 128, 0, stream>>>(A, W2, nullptr, B, N);
    k_self<<<gdiv((long long)N * 16, 256), 256, 0, stream>>>(B, dinv, A, N, 64);
    k_edges<<<gdiv((long long)E * 16, 256), 256, 0, stream>>>(B, dinv, src, dst, ew, A, E, 64);
    k_enc_adj<<<gdiv((long long)N * 64, 256), 256, 0, stream>>>(A, b2, B, N);

    // 4) side features -> enc[:,64:96], enc[:,96:128]
    k_enc_gdv<<<gdiv((long long)N * 32, 256), 256, 0, stream>>>(gdv, gdvW, gdvB, B, N);
    k_enc_pr<<<gdiv((long long)N * 32, 256), 256, 0, stream>>>(pr, prW, prB, B, N);

    // 5) encoder MLP: tanh(enc @ enc_W1 + enc_b1) @ enc_W2 + enc_b2
    k_gemm<128, 1><<<gB, 128, 0, stream>>>(B, encW1, encB1, A, N);
    k_gemm<128, 0><<<gB, 128, 0, stream>>>(A, encW2, encB2, out, N);
}
